// WSDModel_88124138979462
// MI455X (gfx1250) — compile-verified, matrix operand order checked by reading, not run on hardware
//
#include <hip/hip_runtime.h>
#include <hip/hip_bf16.h>

typedef __attribute__((ext_vector_type(16))) __bf16 v16bf;
typedef __attribute__((ext_vector_type(8)))  __bf16 v8bf;
typedef __attribute__((ext_vector_type(8)))  float  v8f;
typedef __attribute__((ext_vector_type(4)))  unsigned int u32x4;
typedef __attribute__((ext_vector_type(8)))  int i32x8;
typedef __attribute__((ext_vector_type(4)))  int i32x4;

#define B_   4
#define N_   2048
#define D_   300
#define DP   320      // D padded to multiple of 32 (WMMA K) and 64 (tile N)
#define Y_   20000
#define YP   20032    // Y padded to multiple of 64
#define MAXREL 10
#define LN_EPS 1e-5f

#define KC   320      // K-chunk staged in LDS per barrier phase
#define KCP  328      // LDS row pitch (elements): 656B = 41*16B (b128-aligned),
                      // dword stride 164 = 36 mod 64 -> conflict-free lanes 0..15

// ---------------------------------------------------------------------------
// TDM: stage a 64-row x `chunk`-col slice of Bt (row stride ldb elements,
// bf16) into LDS at byte offset lds_off with row pitch KCP elements.
// tile_dim0 = KCP > tensor_dim0 = chunk: the 8 pad elements per row are
// zero-filled by the TDM OOB rule and never read by compute.
// D# packing per CDNA5 ISA 8.3/8.4 (2D tensor; groups 2/3 zero).
// ---------------------------------------------------------------------------
__device__ __forceinline__ void tdm_stage_b(const __bf16* gptr, int ldb,
                                            int chunk, unsigned lds_off) {
  unsigned long long ga = (unsigned long long)(uintptr_t)gptr;
  u32x4 g0;
  g0[0] = 1u;                                        // count=1, no gather
  g0[1] = lds_off;                                   // lds_addr (bytes)
  g0[2] = (unsigned)(ga & 0xFFFFFFFFull);            // global_addr[31:0]
  g0[3] = (unsigned)((ga >> 32) & 0x1FFFFFFull)      // global_addr[56:32]
        | (2u << 30);                                // type=2 ("image")
  i32x8 g1;
  g1[0] = (int)(1u << 16);                           // data_size=1 (2 bytes)
  g1[1] = (int)(((unsigned)chunk & 0xFFFFu) << 16);  // tensor_dim0[15:0]
  g1[2] = (int)(((unsigned)chunk >> 16)              // tensor_dim0[31:16]
        | (64u << 16));                              // tensor_dim1[15:0] = 64
  g1[3] = (int)((unsigned)KCP << 16);                // tensor_dim1 hi=0 | tile_dim0
  g1[4] = 64;                                        // tile_dim1=64, tile_dim2=0
  g1[5] = ldb;                                       // tensor_dim0_stride[31:0]
  g1[6] = 0;                                         // stride hi / dim1_stride lo
  g1[7] = 0;
  i32x4 gz4 = {0, 0, 0, 0};
  i32x8 gz8 = {0, 0, 0, 0, 0, 0, 0, 0};
  __builtin_amdgcn_tensor_load_to_lds(g0, g1, gz4, gz4, gz8, 0);
  __builtin_amdgcn_s_wait_tensorcnt(0);
}

// ---------------------------------------------------------------------------
// Fragment loads matching V_WMMA 16-bit operand lane layout:
// lane 0-15  : row = lane,    halves 0..7 = K k0..k0+7,  8..15 = K k0+16..k0+23
// lane 16-31 : row = lane-16, halves 0..7 = K k0+8..+15, 8..15 = K k0+24..+31
// caller passes p = base + row*pitch + k0 + ((lane>>4)<<3); we load 2x16B.
// ---------------------------------------------------------------------------
__device__ __forceinline__ v16bf load_frag_g(const __bf16* __restrict__ p) {
  v8bf lo = *reinterpret_cast<const v8bf*>(p);
  v8bf hi = *reinterpret_cast<const v8bf*>(p + 16);
  v16bf r;
#pragma unroll
  for (int i = 0; i < 8; ++i) { r[i] = lo[i]; r[i + 8] = hi[i]; }
  return r;
}
__device__ __forceinline__ v16bf load_frag_s(const __bf16* p) {
  v8bf lo = *reinterpret_cast<const v8bf*>(p);
  v8bf hi = *reinterpret_cast<const v8bf*>(p + 16);
  v16bf r;
#pragma unroll
  for (int i = 0; i < 8; ++i) { r[i] = lo[i]; r[i + 8] = hi[i]; }
  return r;
}

// ---------------------------------------------------------------------------
// WMMA GEMM: C[z] = A[z] * B[z], Bt = B^T row-major. Block = 128 thr (4
// waves). All 4 waves share one 64-column group (blockIdx.x): the B-tile is
// staged once per K-chunk into LDS by the TDM (wave 0), then each wave runs
// ds_load_b128 -> v_wmma over its private 16-row A stripe. All 4 B fragments
// are fetched before the WMMA chain so LDS latency overlaps WMMA issue.
// grid = (ncols/64, M/64, batches); K multiple of 32; Bt rows padded.
// ---------------------------------------------------------------------------
template <bool OUT_BF16>
__global__ void k_gemm(const __bf16* __restrict__ A, long long a_bs, int lda,
                       const __bf16* __restrict__ Bt, long long b_bs, int ldb,
                       void* __restrict__ C, long long c_bs, int ldc,
                       int K, int ncols_real) {
  __shared__ __bf16 Bs[64 * KCP];                    // 41 KB

  const int lane = threadIdx.x & 31;
  const int wave = threadIdx.x >> 5;
  const int tm   = blockIdx.y * 4 + wave;            // 16-row tile index
  const int tn   = blockIdx.x;                       // 64-col tile group
  const int z    = blockIdx.z;

  const __bf16* Ab = A  + (size_t)z * a_bs;
  const __bf16* Bb = Bt + (size_t)z * b_bs + (size_t)(tn * 64) * ldb;

  const int r16   = lane & 15;
  const int khalf = (lane >> 4) << 3;                // 0 or 8
  const size_t arow = (size_t)(tm * 16 + r16) * lda;
  const unsigned lds_off = (unsigned)(uintptr_t)&Bs[0];

  v8f acc[4] = {v8f{}, v8f{}, v8f{}, v8f{}};

  for (int kc = 0; kc < K; kc += KC) {
    const int chunk = (K - kc < KC) ? (K - kc) : KC;
    if (wave == 0) tdm_stage_b(Bb + kc, ldb, chunk, lds_off);
    __syncthreads();

    for (int kk = 0; kk < chunk; kk += 32) {
      const __bf16* ap = Ab + arow + kc + kk + khalf;
      v16bf av = load_frag_g(ap);
      if (kk + 32 < chunk) __builtin_prefetch(ap + 32, 0, 1);
      const __bf16* bs = &Bs[(size_t)r16 * KCP + kk + khalf];
      // Fetch all 4 B fragments first (one ds_load clause), then the WMMA
      // chain: waits become staggered (dscnt 6/4/2/0) instead of 0 each.
      v16bf bv[4];
#pragma unroll
      for (int j = 0; j < 4; ++j)
        bv[j] = load_frag_s(bs + (size_t)(j * 16) * KCP);
#pragma unroll
      for (int j = 0; j < 4; ++j)
        acc[j] = __builtin_amdgcn_wmma_f32_16x16x32_bf16(
            false, av, false, bv[j], (short)0, acc[j], false, false);
    }
    __syncthreads();                                 // before TDM overwrites Bs
  }

  // C layout: VGPR r -> row (tm*16 + r + 8*(lane>>4)), col = tile + (lane&15)
  const int rbase = tm * 16 + ((lane >> 4) << 3);
  const int cbase = tn * 64 + r16;
  if (OUT_BF16) {
    __bf16* Cc = (__bf16*)C + (size_t)z * c_bs;
#pragma unroll
    for (int j = 0; j < 4; ++j) {
      int col = cbase + j * 16;
      if (col < ncols_real) {
#pragma unroll
        for (int r = 0; r < 8; ++r)
          Cc[(size_t)(rbase + r) * ldc + col] = (__bf16)acc[j][r];
      }
    }
  } else {
    float* Cc = (float*)C + (size_t)z * c_bs;
#pragma unroll
    for (int j = 0; j < 4; ++j) {
      int col = cbase + j * 16;
      if (col < ncols_real) {
#pragma unroll
        for (int r = 0; r < 8; ++r)
          Cc[(size_t)(rbase + r) * ldc + col] = acc[j][r];
      }
    }
  }
}

// ---------------------------------------------------------------------------
// Embedding gather: X_f32[B*N][D], X_bf16[B*N][DP] (zero pad), and
// Xt_bf16[b][DP][N] (transposed, for Q_c = A @ X where Bt must be X^T).
// ---------------------------------------------------------------------------
__global__ void k_gather(const int* __restrict__ Ms, const float* __restrict__ Ev,
                         float* __restrict__ Xf, __bf16* __restrict__ Xb,
                         __bf16* __restrict__ Xt) {
  int row = blockIdx.x;           // b*N + n
  int d   = threadIdx.x;          // 0..319
  int b = row / N_, n = row % N_;
  int idx = Ms[row];
  float v = 0.f;
  if (d < D_) {
    v = Ev[(size_t)idx * D_ + d];
    Xf[(size_t)row * D_ + d] = v;
  }
  __bf16 bv = (__bf16)v;
  Xb[(size_t)row * DP + d] = bv;
  Xt[((size_t)b * DP + d) * N_ + n] = bv;
}

// W_A^T and W_O^T in bf16, zero-padded to DP x DP.
__global__ void k_prep_w(const float* __restrict__ WA, const float* __restrict__ WO,
                         __bf16* __restrict__ WAt, __bf16* __restrict__ WOt) {
  int n = blockIdx.x, k = threadIdx.x;   // both 0..319
  __bf16 a = (__bf16)0.f, o = (__bf16)0.f;
  if (n < D_ && k < D_) {
    a = (__bf16)WA[(size_t)k * D_ + n];
    o = (__bf16)WO[(size_t)k * D_ + n];
  }
  WAt[(size_t)n * DP + k] = a;
  WOt[(size_t)n * DP + k] = o;
}

// E_y bf16, padded to YP x DP (E_y^T row-major == E_y itself as Bt).
__global__ void k_prep_ey(const float* __restrict__ Ey, __bf16* __restrict__ Eyb) {
  int y = blockIdx.x, k = threadIdx.x;
  __bf16 v = (__bf16)0.f;
  if (y < Y_ && k < D_) v = (__bf16)Ey[(size_t)y * D_ + k];
  Eyb[(size_t)y * DP + k] = v;
}

// ---------------------------------------------------------------------------
// Fused distance-bias + key-mask + softmax over a row of N_ scores (in place
// in d_out's A region), plus bf16 copy for the Q_c GEMM. 256 thr, 8 elem/thr.
// ---------------------------------------------------------------------------
__global__ void k_softmax(float* __restrict__ Aout, __bf16* __restrict__ Abf,
                          const int* __restrict__ Ms) {
  const int row = blockIdx.x;     // b*N + n
  const int b = row / N_, n = row % N_;
  float* w = Aout + (size_t)row * N_;
  const int t = threadIdx.x;
  __shared__ float red[256];

  float vals[8];
  float mx = -INFINITY;
#pragma unroll
  for (int i = 0; i < 8; ++i) {
    int j = t + i * 256;
    int dd = n - j; if (dd < 0) dd = -dd; if (dd > MAXREL) dd = MAXREL;
    float v = w[j] - (float)dd;
    if (Ms[b * N_ + j] == 0) v = -INFINITY;
    vals[i] = v;
    mx = fmaxf(mx, v);
  }
  red[t] = mx; __syncthreads();
  for (int s = 128; s > 0; s >>= 1) {
    if (t < s) red[t] = fmaxf(red[t], red[t + s]);
    __syncthreads();
  }
  mx = red[0]; __syncthreads();

  float sum = 0.f;
#pragma unroll
  for (int i = 0; i < 8; ++i) { vals[i] = __expf(vals[i] - mx); sum += vals[i]; }
  red[t] = sum; __syncthreads();
  for (int s = 128; s > 0; s >>= 1) {
    if (t < s) red[t] += red[t + s];
    __syncthreads();
  }
  const float inv = 1.f / red[0];
#pragma unroll
  for (int i = 0; i < 8; ++i) {
    int j = t + i * 256;
    float p = vals[i] * inv;
    w[j] = p;
    Abf[(size_t)row * N_ + j] = (__bf16)p;
  }
}

// ---------------------------------------------------------------------------
// Residual + LayerNorm over D_=300; writes Hn bf16 [row][DP] zero-padded.
// ---------------------------------------------------------------------------
__global__ void k_ln(const float* __restrict__ H1, const float* __restrict__ Xf,
                     const float* __restrict__ gamma, const float* __restrict__ beta,
                     __bf16* __restrict__ Hn) {
  const int row = blockIdx.x;     // b*N + n
  const int t = threadIdx.x;      // 256
  __shared__ float red[256];

  float h[2];
  float s = 0.f;
#pragma unroll
  for (int i = 0; i < 2; ++i) {
    int d = t + i * 256;
    float v = 0.f;
    if (d < D_) v = H1[(size_t)row * DP + d] + Xf[(size_t)row * D_ + d];
    h[i] = v;
    s += v;
  }
  red[t] = s; __syncthreads();
  for (int st = 128; st > 0; st >>= 1) { if (t < st) red[t] += red[t + st]; __syncthreads(); }
  const float mu = red[0] * (1.f / D_); __syncthreads();

  float sq = 0.f;
#pragma unroll
  for (int i = 0; i < 2; ++i) {
    int d = t + i * 256;
    if (d < D_) { float c = h[i] - mu; sq += c * c; }
  }
  red[t] = sq; __syncthreads();
  for (int st = 128; st > 0; st >>= 1) { if (t < st) red[t] += red[t + st]; __syncthreads(); }
  const float rstd = rsqrtf(red[0] * (1.f / D_) + LN_EPS);

#pragma unroll
  for (int i = 0; i < 2; ++i) {
    int d = t + i * 256;
    if (d < D_)
      Hn[(size_t)row * DP + d] = (__bf16)((h[i] - mu) * rstd * gamma[d] + beta[d]);
    else if (d < DP)
      Hn[(size_t)row * DP + d] = (__bf16)0.f;
  }
}

// ---------------------------------------------------------------------------
extern "C" void kernel_launch(void* const* d_in, const int* in_sizes, int n_in,
                              void* d_out, int out_size, void* d_ws, size_t ws_size,
                              hipStream_t stream) {
  const int*   Ms    = (const int*)d_in[0];
  const float* Ev    = (const float*)d_in[1];
  const float* Ey    = (const float*)d_in[2];
  const float* WA    = (const float*)d_in[3];
  const float* WO    = (const float*)d_in[4];
  const float* gamma = (const float*)d_in[5];
  const float* beta  = (const float*)d_in[6];

  float* yout = (float*)d_out;                          // [B,N,Y]
  float* Aout = yout + (size_t)B_ * N_ * Y_;            // [B,N,N]

  // Workspace carve-up (256B aligned, +256B tail slack per buffer).
  char* ws = (char*)d_ws;
  size_t off = 0;
  auto carve = [&](size_t bytes) -> void* {
    void* p = ws + off;
    off = (off + bytes + 511) & ~(size_t)255;
    return p;
  };
  const size_t ROWS = (size_t)B_ * N_;                  // 8192
  float*  Xf  = (float*) carve(ROWS * D_ * sizeof(float));
  __bf16* Xb  = (__bf16*)carve(ROWS * DP * sizeof(__bf16));
  __bf16* Xt  = (__bf16*)carve((size_t)B_ * DP * N_ * sizeof(__bf16));
  __bf16* WAt = (__bf16*)carve((size_t)DP * DP * sizeof(__bf16));
  __bf16* WOt = (__bf16*)carve((size_t)DP * DP * sizeof(__bf16));
  __bf16* Eyb = (__bf16*)carve((size_t)YP * DP * sizeof(__bf16));
  __bf16* QW  = (__bf16*)carve(ROWS * DP * sizeof(__bf16));
  __bf16* Abf = (__bf16*)carve((size_t)B_ * N_ * N_ * sizeof(__bf16));
  __bf16* Qc  = (__bf16*)carve(ROWS * DP * sizeof(__bf16));
  float*  H1  = (float*) carve(ROWS * DP * sizeof(float));
  __bf16* Hn  = (__bf16*)carve(ROWS * DP * sizeof(__bf16));

  // 1) operand prep
  k_gather <<<dim3(ROWS), dim3(DP), 0, stream>>>(Ms, Ev, Xf, Xb, Xt);
  k_prep_w <<<dim3(DP),   dim3(DP), 0, stream>>>(WA, WO, WAt, WOt);
  k_prep_ey<<<dim3(YP),   dim3(DP), 0, stream>>>(Ey, Eyb);

  // 2) QW = X @ W_A         (M=8192, N=320, K=320) -> bf16
  k_gemm<true><<<dim3(DP / 64, ROWS / 64, 1), dim3(128), 0, stream>>>(
      Xb, 0, DP, WAt, 0, DP, QW, 0, DP, DP, DP);

  // 3) S = QW @ X^T per batch (M=2048, N=2048, K=320) -> fp32 into d_out A region
  k_gemm<false><<<dim3(N_ / 64, N_ / 64, B_), dim3(128), 0, stream>>>(
      QW, (long long)N_ * DP, DP, Xb, (long long)N_ * DP, DP,
      Aout, (long long)N_ * N_, N_, DP, N_);

  // 4) bias + mask + softmax (in place) + bf16 copy
  k_softmax<<<dim3(ROWS), dim3(256), 0, stream>>>(Aout, Abf, Ms);

  // 5) Q_c = A @ X per batch (M=2048, N=320, K=2048) -> bf16
  k_gemm<true><<<dim3(DP / 64, N_ / 64, B_), dim3(128), 0, stream>>>(
      Abf, (long long)N_ * N_, N_, Xt, (long long)DP * N_, N_,
      Qc, (long long)N_ * DP, DP, N_, DP);

  // 6) H1 = Q_c @ W_O        (M=8192, N=320, K=320) -> fp32
  k_gemm<false><<<dim3(DP / 64, ROWS / 64, 1), dim3(128), 0, stream>>>(
      Qc, 0, DP, WOt, 0, DP, H1, 0, DP, DP, DP);

  // 7) residual + LayerNorm -> Hn bf16
  k_ln<<<dim3(ROWS), dim3(256), 0, stream>>>(H1, Xf, gamma, beta, Hn);

  // 8) logits = Hn @ E_y^T   (M=8192, N=20032(pad), K=320) -> fp32 d_out
  k_gemm<false><<<dim3(YP / 64, ROWS / 64, 1), dim3(128), 0, stream>>>(
      Hn, 0, DP, Eyb, 0, DP, yout, 0, Y_, DP, Y_);
}
